// DecisionTransformer_24489903522235
// MI455X (gfx1250) — matrix operand order repaired and checked by reading, not compiled, not run on hardware
//
#include <hip/hip_runtime.h>
#include <cstddef>

// ---------------------------------------------------------------------------
// CDNA5 (gfx1250) fused Decision-Transformer + TreeLSTM.
// All GEMMs use v_wmma_f32_16x16x32_bf16 (bf16 A/B, fp32 accumulate).
// ---------------------------------------------------------------------------

typedef __attribute__((ext_vector_type(16))) __bf16 bvec16;
typedef __attribute__((ext_vector_type(8)))  float  fvec8;

#define BT   32
#define TT   64
#define NN   8
#define ROWS 16384          // B*T*N
#define SEQS 256            // B*N
#define SLEN 192            // 3*T
#define EMBD 128
#define DFFD 2048

__device__ __forceinline__ fvec8 wmma_bf16(bvec16 a, bvec16 b, fvec8 c) {
  return __builtin_amdgcn_wmma_f32_16x16x32_bf16(false, a, false, b, (short)0, c, false, false);
}

// B fragment from col-major bf16 matrix Wt[N][Kpad] (LDS or global).
// lane L, element e: k = kbase + 16*(L>>4) + e ; n = ncol (absolute column).
__device__ __forceinline__ bvec16 load_bfrag(const __bf16* Wt, int Kpad, int ncol,
                                             int kbase, int lane) {
  const __bf16* p = Wt + (size_t)ncol * Kpad + kbase + ((lane >> 4) << 4);
  bvec16 b;
#pragma unroll
  for (int e = 0; e < 16; ++e) b[e] = p[e];
  return b;
}

// A fragment via value getter (row,k)->float.  row = L&15,
// k = kbase + 16*(L>>4) + e
template <typename F>
__device__ __forceinline__ bvec16 make_afrag(int lane, int kbase, F getv) {
  int row = lane & 15;
  int kb  = kbase + ((lane >> 4) << 4);
  bvec16 a;
#pragma unroll
  for (int e = 0; e < 16; ++e) {
    int k = kb + e;
    a[e]  = (__bf16)getv(row, k);
  }
  return a;
}

// branch-free fast transcendentals (keep EXEC uniform around WMMA streams)
__device__ __forceinline__ float sigm(float x) {
  return __builtin_amdgcn_rcpf(1.0f + __expf(-x));
}
__device__ __forceinline__ float tanh_fast(float x) {
  float xc = fminf(fmaxf(x, -15.f), 15.f);
  float e  = __expf(2.f * xc);
  return 1.f - 2.f * __builtin_amdgcn_rcpf(e + 1.f);
}
__device__ __forceinline__ float gelu_(float x) {
  return 0.5f * x * (1.0f + erff(x * 0.7071067811865475f));
}
// sum across the 16 lanes of a half-wave (rows of a WMMA C/D tile)
__device__ __forceinline__ float half16_sum(float v) {
  v += __shfl_xor(v, 1, 32);
  v += __shfl_xor(v, 2, 32);
  v += __shfl_xor(v, 4, 32);
  v += __shfl_xor(v, 8, 32);
  return v;
}

// ---------------------------------------------------------------------------
// Fused TreeLSTM.  Fixed topology: node0 -> nodes1..4 -> leaves5..20.
// One wave per tree, 8 trees/block.  Stacked weights in LDS (col-major bf16):
//   Wc[n][k], n<384 iou-cols / n>=384 f-cols ; k<12 tw, 12..139 tu, pad to 160
// ---------------------------------------------------------------------------
__global__ void k_tree(const float* __restrict__ forest, const float* __restrict__ tw_iou,
                       const float* __restrict__ tb_iou, const float* __restrict__ tu_iou,
                       const float* __restrict__ tw_f, const float* __restrict__ tb_f,
                       const float* __restrict__ tu_f, __bf16* __restrict__ root_h) {
  extern __shared__ char smem[];
  __bf16* Wc    = (__bf16*)(smem);                   // 512*160  (163840 B)
  float*  xsb   = (float*)(smem + 163840);           // 8*21*12  (8064 B)
  float*  hsumb = (float*)(smem + 163840 + 8064);    // 8*4*128  (16384 B)
  float*  csumb = (float*)(smem + 188288);           // 8*4*128  (16384 B)
  __bf16* hlb   = (__bf16*)(smem + 204672);          // 8*16*128 (32768 B)
  __bf16* clb   = (__bf16*)(smem + 237440);          // 8*16*128 (32768 B)
  __bf16* hmb   = (__bf16*)(smem + 270208);          // 8*4*128  (8192 B)
  __bf16* cmb   = (__bf16*)(smem + 278400);          // 8*4*128  (8192 B)  end=286592

  for (int i = threadIdx.x; i < 512 * 160; i += 256) {
    int n = i / 160, k = i % 160;
    float v = 0.f;
    if (n < 384) {
      if (k < 12) v = tw_iou[k * 384 + n];
      else if (k < 140) v = tu_iou[(k - 12) * 384 + n];
    } else {
      int nc = n - 384;
      if (k < 12) v = tw_f[k * 128 + nc];
      else if (k < 140) v = tu_f[(k - 12) * 128 + nc];
    }
    Wc[i] = (__bf16)v;
  }
  int wave = threadIdx.x >> 5, lane = threadIdx.x & 31;
  int tree = blockIdx.x * 8 + wave;
  const float* xin = forest + (size_t)tree * (21 * 12);
  float* xw = xsb + wave * 252;
  for (int i = lane; i < 252; i += 32) xw[i] = xin[i];
  __syncthreads();

  __bf16* hl = hlb + wave * 16 * 128;
  __bf16* cl = clb + wave * 16 * 128;
  float*  hs = hsumb + wave * 4 * 128;
  float*  cs = csumb + wave * 4 * 128;
  __bf16* hm = hmb + wave * 4 * 128;
  __bf16* cm = cmb + wave * 4 * 128;

  // ---- level 0: leaves (h_sum == 0, only k<12 active -> single k-step) ----
  for (int nt = 0; nt < 8; ++nt) {
    fvec8 ai = {}, ao = {}, au = {};
    int nci = nt * 16 + (lane & 15);
    bvec16 af = make_afrag(lane, 0, [&](int row, int k) -> float {
      return (k < 12) ? xw[(5 + row) * 12 + k] : 0.f;
    });
    ai = wmma_bf16(af, load_bfrag(Wc, 160, nci, 0, lane), ai);
    ao = wmma_bf16(af, load_bfrag(Wc, 160, nci + 128, 0, lane), ao);
    au = wmma_bf16(af, load_bfrag(Wc, 160, nci + 256, 0, lane), au);
    float bi = tb_iou[nci], bo = tb_iou[nci + 128], bu = tb_iou[nci + 256];
#pragma unroll
    for (int r = 0; r < 8; ++r) {
      int leaf = r + ((lane >> 4) << 3);
      float iv = sigm(ai[r] + bi), ov = sigm(ao[r] + bo), uv = tanh_fast(au[r] + bu);
      float cv = iv * uv;
      hl[leaf * 128 + nci] = (__bf16)(ov * tanh_fast(cv));
      cl[leaf * 128 + nci] = (__bf16)cv;
    }
  }
  __syncthreads();
  // ---- h_sum for mid nodes ----
  for (int i = lane; i < 512; i += 32) {
    int p = i >> 7, c = i & 127;
    hs[i] = (float)hl[(4 * p + 0) * 128 + c] + (float)hl[(4 * p + 1) * 128 + c] +
            (float)hl[(4 * p + 2) * 128 + c] + (float)hl[(4 * p + 3) * 128 + c];
  }
  __syncthreads();
  // ---- level 1 forget-gates: 16 edge rows, f = sig([x_par|h_child] @ Wf) ----
  for (int nt = 0; nt < 8; ++nt) {
    fvec8 acc = {};
    int ncol = 384 + nt * 16 + (lane & 15);
#pragma unroll
    for (int ks = 0; ks < 5; ++ks) {
      bvec16 af = make_afrag(lane, ks * 32, [&](int row, int k) -> float {
        if (k < 12) return xw[(1 + (row >> 2)) * 12 + k];
        if (k < 140) return (float)hl[row * 128 + (k - 12)];
        return 0.f;
      });
      acc = wmma_bf16(af, load_bfrag(Wc, 160, ncol, ks * 32, lane), acc);
    }
    int colf = nt * 16 + (lane & 15);
    float bf = tb_f[colf];
#pragma unroll
    for (int pl = 0; pl < 2; ++pl) {
      int p = pl + ((lane >> 4) << 1);
      float s = 0.f;
#pragma unroll
      for (int j = 0; j < 4; ++j)
        s += sigm(acc[pl * 4 + j] + bf) * (float)cl[(4 * p + j) * 128 + colf];
      cs[p * 128 + colf] = s;
    }
  }
  __syncthreads();
  // ---- level 1 iou for mid nodes (rows 0..3) ----
  for (int nt = 0; nt < 8; ++nt) {
    fvec8 ai = {}, ao = {}, au = {};
    int nci = nt * 16 + (lane & 15);
#pragma unroll
    for (int ks = 0; ks < 5; ++ks) {
      bvec16 af = make_afrag(lane, ks * 32, [&](int row, int k) -> float {
        if (row >= 4) return 0.f;
        if (k < 12) return xw[(1 + row) * 12 + k];
        if (k < 140) return hs[row * 128 + (k - 12)];
        return 0.f;
      });
      ai = wmma_bf16(af, load_bfrag(Wc, 160, nci, ks * 32, lane), ai);
      ao = wmma_bf16(af, load_bfrag(Wc, 160, nci + 128, ks * 32, lane), ao);
      au = wmma_bf16(af, load_bfrag(Wc, 160, nci + 256, ks * 32, lane), au);
    }
    float bi = tb_iou[nci], bo = tb_iou[nci + 128], bu = tb_iou[nci + 256];
    if (lane < 16) {
#pragma unroll
      for (int r = 0; r < 4; ++r) {
        float iv = sigm(ai[r] + bi), ov = sigm(ao[r] + bo), uv = tanh_fast(au[r] + bu);
        float cv = iv * uv + cs[r * 128 + nci];
        hm[r * 128 + nci] = (__bf16)(ov * tanh_fast(cv));
        cm[r * 128 + nci] = (__bf16)cv;
      }
    }
  }
  __syncthreads();
  // ---- root forget gates (4 edge rows) ----
  float csr[8];
#pragma unroll
  for (int nt = 0; nt < 8; ++nt) {
    fvec8 acc = {};
    int ncol = 384 + nt * 16 + (lane & 15);
#pragma unroll
    for (int ks = 0; ks < 5; ++ks) {
      bvec16 af = make_afrag(lane, ks * 32, [&](int row, int k) -> float {
        if (row >= 4) return 0.f;
        if (k < 12) return xw[k];
        if (k < 140) return (float)hm[row * 128 + (k - 12)];
        return 0.f;
      });
      acc = wmma_bf16(af, load_bfrag(Wc, 160, ncol, ks * 32, lane), acc);
    }
    int colf = nt * 16 + (lane & 15);
    float bf = tb_f[colf], s = 0.f;
#pragma unroll
    for (int r = 0; r < 4; ++r) s += sigm(acc[r] + bf) * (float)cm[r * 128 + colf];
    csr[nt] = s;  // valid in lanes < 16
  }
  // ---- root iou ----
#pragma unroll
  for (int nt = 0; nt < 8; ++nt) {
    fvec8 ai = {}, ao = {}, au = {};
    int nci = nt * 16 + (lane & 15);
#pragma unroll
    for (int ks = 0; ks < 5; ++ks) {
      bvec16 af = make_afrag(lane, ks * 32, [&](int row, int k) -> float {
        if (row != 0) return 0.f;
        if (k < 12) return xw[k];
        if (k < 140) {
          int c = k - 12;
          return (float)hm[c] + (float)hm[128 + c] + (float)hm[256 + c] + (float)hm[384 + c];
        }
        return 0.f;
      });
      ai = wmma_bf16(af, load_bfrag(Wc, 160, nci, ks * 32, lane), ai);
      ao = wmma_bf16(af, load_bfrag(Wc, 160, nci + 128, ks * 32, lane), ao);
      au = wmma_bf16(af, load_bfrag(Wc, 160, nci + 256, ks * 32, lane), au);
    }
    if (lane < 16) {
      float iv = sigm(ai[0] + tb_iou[nci]);
      float ov = sigm(ao[0] + tb_iou[nci + 128]);
      float uv = tanh_fast(au[0] + tb_iou[nci + 256]);
      float cv = iv * uv + csr[nt];
      root_h[(size_t)tree * 128 + nci] = (__bf16)(ov * tanh_fast(cv));
    }
  }
}

// ---------------------------------------------------------------------------
// attr MLP: gelu(gelu(attr @ W1 + b1) @ W2 + b2)   (16384 x 83 -> 128)
// ---------------------------------------------------------------------------
__global__ void k_attr(const float* __restrict__ ain, const float* __restrict__ w1,
                       const float* __restrict__ b1, const float* __restrict__ w2,
                       const float* __restrict__ b2, __bf16* __restrict__ attr_h) {
  extern __shared__ char smem[];
  __bf16* W1t = (__bf16*)smem;                    // [256][96]
  __bf16* W2t = (__bf16*)(smem + 49152);          // [128][256]
  __bf16* hid = (__bf16*)(smem + 49152 + 65536);  // [128][256]
  for (int i = threadIdx.x; i < 256 * 96; i += 256) {
    int n = i / 96, k = i % 96;
    W1t[i] = (__bf16)(k < 83 ? w1[k * 256 + n] : 0.f);
  }
  for (int i = threadIdx.x; i < 128 * 256; i += 256) {
    int n = i >> 8, k = i & 255;
    W2t[i] = (__bf16)w2[k * 128 + n];
  }
  __syncthreads();
  int wave = threadIdx.x >> 5, lane = threadIdx.x & 31;
  int row0 = blockIdx.x * 128 + wave * 16;
  const float* A = ain + (size_t)row0 * 83;
  __bf16* hw = hid + wave * 16 * 256;
  {
    fvec8 acc[16];
#pragma unroll
    for (int nt = 0; nt < 16; ++nt)
#pragma unroll
      for (int e = 0; e < 8; ++e) acc[nt][e] = 0.f;
#pragma unroll
    for (int ks = 0; ks < 3; ++ks) {
      bvec16 af = make_afrag(lane, ks * 32, [&](int row, int k) -> float {
        return (k < 83) ? A[row * 83 + k] : 0.f;
      });
#pragma unroll
      for (int nt = 0; nt < 16; ++nt)
        acc[nt] = wmma_bf16(af, load_bfrag(W1t, 96, nt * 16 + (lane & 15), ks * 32, lane), acc[nt]);
    }
#pragma unroll
    for (int nt = 0; nt < 16; ++nt) {
      int ncol = nt * 16 + (lane & 15);
      float bb = b1[ncol];
#pragma unroll
      for (int r = 0; r < 8; ++r) {
        int rg = r + ((lane >> 4) << 3);
        hw[rg * 256 + ncol] = (__bf16)gelu_(acc[nt][r] + bb);
      }
    }
  }
  __syncthreads();
  {
    fvec8 acc[8];
#pragma unroll
    for (int nt = 0; nt < 8; ++nt)
#pragma unroll
      for (int e = 0; e < 8; ++e) acc[nt][e] = 0.f;
#pragma unroll
    for (int ks = 0; ks < 8; ++ks) {
      bvec16 af = make_afrag(lane, ks * 32,
                             [&](int row, int k) -> float { return (float)hw[row * 256 + k]; });
#pragma unroll
      for (int nt = 0; nt < 8; ++nt)
        acc[nt] =
            wmma_bf16(af, load_bfrag(W2t, 256, nt * 16 + (lane & 15), ks * 32, lane), acc[nt]);
    }
#pragma unroll
    for (int nt = 0; nt < 8; ++nt) {
      int ncol = nt * 16 + (lane & 15);
      float bb = b2[ncol];
#pragma unroll
      for (int r = 0; r < 8; ++r) {
        int rg = r + ((lane >> 4) << 3);
        attr_h[(size_t)(row0 + rg) * 128 + ncol] = (__bf16)gelu_(acc[nt][r] + bb);
      }
    }
  }
}

// ---------------------------------------------------------------------------
// state = [root | attr] @ proj_w + proj_b         (16384 x 256 -> 256)
// ---------------------------------------------------------------------------
__global__ void k_proj(const __bf16* __restrict__ root_h, const __bf16* __restrict__ attr_h,
                       const float* __restrict__ pw, const float* __restrict__ pb,
                       __bf16* __restrict__ state) {
  extern __shared__ char smem[];
  __bf16* Pt = (__bf16*)smem;  // [256][256]
  for (int i = threadIdx.x; i < 256 * 256; i += 256) {
    int n = i >> 8, k = i & 255;
    Pt[i] = (__bf16)pw[k * 256 + n];
  }
  __syncthreads();
  int wave = threadIdx.x >> 5, lane = threadIdx.x & 31;
  int row0 = blockIdx.x * 128 + wave * 16;
  fvec8 acc[16];
#pragma unroll
  for (int nt = 0; nt < 16; ++nt)
#pragma unroll
    for (int e = 0; e < 8; ++e) acc[nt][e] = 0.f;
#pragma unroll
  for (int ks = 0; ks < 8; ++ks) {
    bvec16 af = make_afrag(lane, ks * 32, [&](int row, int k) -> float {
      size_t rr = (size_t)(row0 + row);
      return (k < 128) ? (float)root_h[rr * 128 + k] : (float)attr_h[rr * 128 + (k - 128)];
    });
#pragma unroll
    for (int nt = 0; nt < 16; ++nt)
      acc[nt] = wmma_bf16(af, load_bfrag(Pt, 256, nt * 16 + (lane & 15), ks * 32, lane), acc[nt]);
  }
#pragma unroll
  for (int nt = 0; nt < 16; ++nt) {
    int ncol = nt * 16 + (lane & 15);
    float bb = pb[ncol];
#pragma unroll
    for (int r = 0; r < 8; ++r) {
      int rg = r + ((lane >> 4) << 3);
      state[(size_t)(row0 + rg) * 256 + ncol] = (__bf16)(acc[nt][r] + bb);
    }
  }
}

// ---------------------------------------------------------------------------
// s = state @ st_w + st_b  -> scattered into interleaved token buffer (pos 3t+1)
// ---------------------------------------------------------------------------
__global__ void k_s(const __bf16* __restrict__ state, const float* __restrict__ sw,
                    const float* __restrict__ sb, float* __restrict__ x) {
  extern __shared__ char smem[];
  __bf16* St = (__bf16*)smem;  // [128][256]
  for (int i = threadIdx.x; i < 128 * 256; i += 256) {
    int n = i >> 8, k = i & 255;
    St[i] = (__bf16)sw[k * 128 + n];
  }
  __syncthreads();
  int wave = threadIdx.x >> 5, lane = threadIdx.x & 31;
  int row0 = blockIdx.x * 128 + wave * 16;
  fvec8 acc[8];
#pragma unroll
  for (int nt = 0; nt < 8; ++nt)
#pragma unroll
    for (int e = 0; e < 8; ++e) acc[nt][e] = 0.f;
#pragma unroll
  for (int ks = 0; ks < 8; ++ks) {
    bvec16 af = make_afrag(lane, ks * 32, [&](int row, int k) -> float {
      return (float)state[(size_t)(row0 + row) * 256 + k];
    });
#pragma unroll
    for (int nt = 0; nt < 8; ++nt)
      acc[nt] = wmma_bf16(af, load_bfrag(St, 256, nt * 16 + (lane & 15), ks * 32, lane), acc[nt]);
  }
#pragma unroll
  for (int nt = 0; nt < 8; ++nt) {
    int ncol = nt * 16 + (lane & 15);
    float bb = sb[ncol];
#pragma unroll
    for (int r = 0; r < 8; ++r) {
      int rg  = r + ((lane >> 4) << 3);
      int row = row0 + rg;
      int b = row >> 9, t = (row >> 3) & 63, n = row & 7;
      x[(((size_t)(b * 8 + n)) * SLEN + 3 * t + 1) * 128 + ncol] = acc[nt][r] + bb;
    }
  }
}

// r and a tokens (pos 3t, 3t+2)
__global__ void k_ra(const float* __restrict__ rtgs, const int* __restrict__ actions,
                     const float* __restrict__ act_emb, const float* __restrict__ rw,
                     const float* __restrict__ rb, float* __restrict__ x) {
  int idx = blockIdx.x * 256 + threadIdx.x;  // ROWS*128
  int row = idx >> 7, d = idx & 127;
  int b = row >> 9, t = (row >> 3) & 63, n = row & 7;
  size_t base = ((size_t)(b * 8 + n)) * SLEN + 3 * t;
  x[base * 128 + d]       = rtgs[row] * rw[d] + rb[d];
  x[(base + 2) * 128 + d] = act_emb[actions[row] * 128 + d];
}

// qkv = x @ in_w + in_b   (49152 x 128 -> 384, bf16 out)
__global__ void k_qkv(const float* __restrict__ x, const float* __restrict__ w,
                      const float* __restrict__ b, __bf16* __restrict__ qkv) {
  extern __shared__ char smem[];
  __bf16* Wt = (__bf16*)smem;  // [384][128]
  for (int i = threadIdx.x; i < 384 * 128; i += 256) {
    int n = i >> 7, k = i & 127;
    Wt[i] = (__bf16)w[k * 384 + n];
  }
  __syncthreads();
  int wave = threadIdx.x >> 5, lane = threadIdx.x & 31;
  int row0 = blockIdx.x * 128 + wave * 16;
  const float* A = x + (size_t)row0 * 128;
  __builtin_prefetch(A, 0, 1);
#pragma unroll
  for (int g = 0; g < 2; ++g) {
    fvec8 acc[12];
#pragma unroll
    for (int nt = 0; nt < 12; ++nt)
#pragma unroll
      for (int e = 0; e < 8; ++e) acc[nt][e] = 0.f;
#pragma unroll
    for (int ks = 0; ks < 4; ++ks) {
      bvec16 af = make_afrag(lane, ks * 32,
                             [&](int row, int k) -> float { return A[row * 128 + k]; });
#pragma unroll
      for (int nt = 0; nt < 12; ++nt)
        acc[nt] = wmma_bf16(
            af, load_bfrag(Wt, 128, (g * 12 + nt) * 16 + (lane & 15), ks * 32, lane), acc[nt]);
    }
#pragma unroll
    for (int nt = 0; nt < 12; ++nt) {
      int ncol = (g * 12 + nt) * 16 + (lane & 15);
      float bb = b[ncol];
#pragma unroll
      for (int r = 0; r < 8; ++r) {
        int rg = r + ((lane >> 4) << 3);
        qkv[(size_t)(row0 + rg) * 384 + ncol] = (__bf16)(acc[nt][r] + bb);
      }
    }
  }
}

// attention for one (seq, head): softmax(q k^T / sqrt(32)) v
__global__ void k_attn(const __bf16* __restrict__ qkv, __bf16* __restrict__ o) {
  extern __shared__ char smem[];
  float*  sc = (float*)smem;                                  // [192][192]
  __bf16* P  = (__bf16*)(smem + SLEN * SLEN * 4);             // [192][192]
  __bf16* Vt = (__bf16*)(smem + SLEN * SLEN * 4 + SLEN * SLEN * 2);  // [32][192]
  int seq = blockIdx.x >> 2, head = blockIdx.x & 3;
  const __bf16* q  = qkv + (size_t)seq * SLEN * 384 + head * 32;
  const __bf16* km = q + 128;
  const __bf16* v  = q + 256;
  for (int i = threadIdx.x; i < 32 * SLEN; i += 256) {
    int d = i / SLEN, tok = i % SLEN;
    Vt[d * SLEN + tok] = v[(size_t)tok * 384 + d];
  }
  __syncthreads();
  int wave = threadIdx.x >> 5, lane = threadIdx.x & 31;
  for (int tile = wave; tile < 144; tile += 8) {
    int mt = tile / 12, nt = tile % 12;
    fvec8 acc = {};
    bvec16 af = make_afrag(lane, 0, [&](int row, int k) -> float {
      return (float)q[(size_t)(mt * 16 + row) * 384 + k];
    });
    bvec16 bf;
    {
      int n = lane & 15, kb = (lane >> 4) << 4;
      const __bf16* p = km + (size_t)(nt * 16 + n) * 384 + kb;
#pragma unroll
      for (int e = 0; e < 16; ++e) bf[e] = p[e];
    }
    acc = wmma_bf16(af, bf, acc);
#pragma unroll
    for (int r = 0; r < 8; ++r) {
      int rg = r + ((lane >> 4) << 3);
      sc[(mt * 16 + rg) * SLEN + nt * 16 + (lane & 15)] = acc[r] * 0.17677669529663687f;
    }
  }
  __syncthreads();
  for (int row = threadIdx.x; row < SLEN; row += 256) {
    float m = -1e30f;
    for (int j = 0; j < SLEN; ++j) m = fmaxf(m, sc[row * SLEN + j]);
    float s = 0.f;
    for (int j = 0; j < SLEN; ++j) {
      float e = __expf(sc[row * SLEN + j] - m);
      sc[row * SLEN + j] = e;
      s += e;
    }
    float inv = __builtin_amdgcn_rcpf(s);
    for (int j = 0; j < SLEN; ++j) P[row * SLEN + j] = (__bf16)(sc[row * SLEN + j] * inv);
  }
  __syncthreads();
  for (int job = wave; job < 24; job += 8) {
    int mt = job >> 1, nt = job & 1;
    fvec8 acc = {};
#pragma unroll
    for (int ks = 0; ks < 6; ++ks) {
      bvec16 af = make_afrag(lane, ks * 32, [&](int row, int k) -> float {
        return (float)P[(mt * 16 + row) * SLEN + k];
      });
      acc = wmma_bf16(af, load_bfrag(Vt, SLEN, nt * 16 + (lane & 15), ks * 32, lane), acc);
    }
#pragma unroll
    for (int r = 0; r < 8; ++r) {
      int rg = r + ((lane >> 4) << 3);
      o[((size_t)seq * SLEN + mt * 16 + rg) * 128 + head * 32 + nt * 16 + (lane & 15)] =
          (__bf16)acc[r];
    }
  }
}

// x = LN1(x + o @ out_w + out_b)    (in-place on x)
__global__ void k_oln(const __bf16* __restrict__ o, const float* __restrict__ ow,
                      const float* __restrict__ ob, const float* __restrict__ g,
                      const float* __restrict__ gb, float* __restrict__ x) {
  extern __shared__ char smem[];
  __bf16* Ot = (__bf16*)smem;  // [128][128]
  for (int i = threadIdx.x; i < 128 * 128; i += 256) {
    int n = i >> 7, k = i & 127;
    Ot[i] = (__bf16)ow[k * 128 + n];
  }
  __syncthreads();
  int wave = threadIdx.x >> 5, lane = threadIdx.x & 31;
  int row0 = blockIdx.x * 128 + wave * 16;
  fvec8 acc[8];
#pragma unroll
  for (int nt = 0; nt < 8; ++nt)
#pragma unroll
    for (int e = 0; e < 8; ++e) acc[nt][e] = 0.f;
#pragma unroll
  for (int ks = 0; ks < 4; ++ks) {
    bvec16 af = make_afrag(lane, ks * 32, [&](int row, int k) -> float {
      return (float)o[(size_t)(row0 + row) * 128 + k];
    });
#pragma unroll
    for (int nt = 0; nt < 8; ++nt)
      acc[nt] = wmma_bf16(af, load_bfrag(Ot, 128, nt * 16 + (lane & 15), ks * 32, lane), acc[nt]);
  }
#pragma unroll
  for (int r = 0; r < 8; ++r) {
    int rg = r + ((lane >> 4) << 3);
    size_t ro = (size_t)(row0 + rg) * 128;
    float v[8], s = 0.f;
#pragma unroll
    for (int nt = 0; nt < 8; ++nt) {
      int c = nt * 16 + (lane & 15);
      v[nt] = acc[nt][r] + ob[c] + x[ro + c];
      s += v[nt];
    }
    float mean = half16_sum(s) * (1.f / 128.f);
    float s2 = 0.f;
#pragma unroll
    for (int nt = 0; nt < 8; ++nt) {
      float d = v[nt] - mean;
      s2 += d * d;
    }
    float rs = rsqrtf(half16_sum(s2) * (1.f / 128.f) + 1e-5f);
#pragma unroll
    for (int nt = 0; nt < 8; ++nt) {
      int c = nt * 16 + (lane & 15);
      x[ro + c] = (v[nt] - mean) * rs * g[c] + gb[c];
    }
  }
}

// x = LN2(x + relu(x @ ff_w1 + b1) @ ff_w2 + b2)    (16 rows per block)
__global__ void k_ff(float* __restrict__ x, const __bf16* __restrict__ w1t,
                     const float* __restrict__ b1, const __bf16* __restrict__ w2t,
                     const float* __restrict__ b2, const float* __restrict__ g,
                     const float* __restrict__ gb) {
  extern __shared__ char smem[];
  __bf16* hid  = (__bf16*)smem;                   // [16][2048]
  float*  outb = (float*)(smem + 16 * 2048 * 2);  // [16][128]
  int wave = threadIdx.x >> 5, lane = threadIdx.x & 31;
  int row0 = blockIdx.x * 16;
  const float* A = x + (size_t)row0 * 128;
  __builtin_prefetch(A, 0, 1);
  {
    fvec8 acc[16];
#pragma unroll
    for (int i = 0; i < 16; ++i)
#pragma unroll
      for (int e = 0; e < 8; ++e) acc[i][e] = 0.f;
#pragma unroll
    for (int ks = 0; ks < 4; ++ks) {
      bvec16 af = make_afrag(lane, ks * 32,
                             [&](int row, int k) -> float { return A[row * 128 + k]; });
#pragma unroll
      for (int i = 0; i < 16; ++i) {
        int ncol = (wave * 16 + i) * 16 + (lane & 15);
        acc[i] = wmma_bf16(af, load_bfrag(w1t, 128, ncol, ks * 32, lane), acc[i]);
      }
    }
#pragma unroll
    for (int i = 0; i < 16; ++i) {
      int ncol = (wave * 16 + i) * 16 + (lane & 15);
      float bb = b1[ncol];
#pragma unroll
      for (int r = 0; r < 8; ++r) {
        int rg  = r + ((lane >> 4) << 3);
        float v = acc[i][r] + bb;
        hid[rg * 2048 + ncol] = (__bf16)(v > 0.f ? v : 0.f);
      }
    }
  }
  __syncthreads();
  {
    fvec8 acc = {};
    int ncol = wave * 16 + (lane & 15);
    for (int ks = 0; ks < 64; ++ks) {
      bvec16 af = make_afrag(lane, ks * 32,
                             [&](int row, int k) -> float { return (float)hid[row * 2048 + k]; });
      acc = wmma_bf16(af, load_bfrag(w2t, 2048, ncol, ks * 32, lane), acc);
    }
    float bb = b2[ncol];
#pragma unroll
    for (int r = 0; r < 8; ++r) {
      int rg = r + ((lane >> 4) << 3);
      outb[rg * 128 + ncol] = acc[r] + bb + A[rg * 128 + ncol];
    }
  }
  __syncthreads();
  {
    int row = threadIdx.x >> 4, lc = threadIdx.x & 15;
    float v[8], s = 0.f;
#pragma unroll
    for (int j = 0; j < 8; ++j) {
      v[j] = outb[row * 128 + lc + 16 * j];
      s += v[j];
    }
    float mean = half16_sum(s) * (1.f / 128.f);
    float s2 = 0.f;
#pragma unroll
    for (int j = 0; j < 8; ++j) {
      float d = v[j] - mean;
      s2 += d * d;
    }
    float rs = rsqrtf(half16_sum(s2) * (1.f / 128.f) + 1e-5f);
#pragma unroll
    for (int j = 0; j < 8; ++j) {
      int c = lc + 16 * j;
      x[(size_t)(row0 + row) * 128 + c] = (v[j] - mean) * rs * g[c] + gb[c];
    }
  }
}

// generic f32 (K x N) -> bf16 col-major [N][Kpad]
__global__ void k_prep(const float* __restrict__ src, __bf16* __restrict__ dst, int K, int Nc,
                       int Kpad) {
  int i = blockIdx.x * 256 + threadIdx.x;
  if (i >= Nc * Kpad) return;
  int n = i / Kpad, k = i % Kpad;
  dst[i] = (__bf16)(k < K ? src[(size_t)k * Nc + n] : 0.f);
}

// head: out = gelu(x_last @ pred_w1 + b1) @ pred_w2 + b2   (256 x 5)
__global__ void k_head(const float* __restrict__ x, const float* __restrict__ w1,
                       const float* __restrict__ b1, const float* __restrict__ w2,
                       const float* __restrict__ b2, float* __restrict__ out) {
  extern __shared__ char smem[];
  __bf16* W1t = (__bf16*)smem;                   // [128][128]
  __bf16* W2t = (__bf16*)(smem + 32768);         // [16][128] (cols >=5 zero)
  __bf16* hid = (__bf16*)(smem + 32768 + 4096);  // [256][128]
  for (int i = threadIdx.x; i < 128 * 128; i += 256) {
    int n = i >> 7, k = i & 127;
    W1t[i] = (__bf16)w1[k * 128 + n];
  }
  for (int i = threadIdx.x; i < 16 * 128; i += 256) {
    int n = i >> 7, k = i & 127;
    W2t[i] = (__bf16)(n < 5 ? w2[k * 5 + n] : 0.f);
  }
  __syncthreads();
  int wave = threadIdx.x >> 5, lane = threadIdx.x & 31;
  for (int mt = wave; mt < 16; mt += 8) {
    fvec8 acc[8];
#pragma unroll
    for (int nt = 0; nt < 8; ++nt)
#pragma unroll
      for (int e = 0; e < 8; ++e) acc[nt][e] = 0.f;
#pragma unroll
    for (int ks = 0; ks < 4; ++ks) {
      bvec16 af = make_afrag(lane, ks * 32, [&](int row, int k) -> float {
        return x[((size_t)(mt * 16 + row) * SLEN + (SLEN - 1)) * 128 + k];
      });
#pragma unroll
      for (int nt = 0; nt < 8; ++nt)
        acc[nt] =
            wmma_bf16(af, load_bfrag(W1t, 128, nt * 16 + (lane & 15), ks * 32, lane), acc[nt]);
    }
#pragma unroll
    for (int nt = 0; nt < 8; ++nt) {
      int ncol = nt * 16 + (lane & 15);
      float bb = b1[ncol];
#pragma unroll
      for (int r = 0; r < 8; ++r) {
        int rg = r + ((lane >> 4) << 3);
        hid[(mt * 16 + rg) * 128 + ncol] = (__bf16)gelu_(acc[nt][r] + bb);
      }
    }
  }
  __syncthreads();
  for (int mt = wave; mt < 16; mt += 8) {
    fvec8 acc = {};
#pragma unroll
    for (int ks = 0; ks < 4; ++ks) {
      bvec16 af = make_afrag(lane, ks * 32, [&](int row, int k) -> float {
        return (float)hid[(mt * 16 + row) * 128 + k];
      });
      acc = wmma_bf16(af, load_bfrag(W2t, 128, lane & 15, ks * 32, lane), acc);
    }
    int col = lane & 15;
    if (col < 5) {
#pragma unroll
      for (int r = 0; r < 8; ++r) {
        int rg = r + ((lane >> 4) << 3);
        out[(mt * 16 + rg) * 5 + col] = acc[r] + b2[col];
      }
    }
  }
}

// ---------------------------------------------------------------------------
extern "C" void kernel_launch(void* const* d_in, const int* in_sizes, int n_in, void* d_out,
                              int out_size, void* d_ws, size_t ws_size, hipStream_t stream) {
  (void)in_sizes; (void)n_in; (void)out_size; (void)ws_size;
  const float* forest  = (const float*)d_in[0];
  const float* aattr   = (const float*)d_in[1];
  const int*   actions = (const int*)d_in[5];
  const float* rtgs    = (const float*)d_in[6];
  const float *tw_iou = (const float*)d_in[8], *tb_iou = (const float*)d_in[9];
  const float *tu_iou = (const float*)d_in[10], *tw_f = (const float*)d_in[11];
  const float *tb_f = (const float*)d_in[12], *tu_f = (const float*)d_in[13];
  const float *aw1 = (const float*)d_in[14], *ab1 = (const float*)d_in[15];
  const float *aw2 = (const float*)d_in[16], *ab2 = (const float*)d_in[17];
  const float *pw = (const float*)d_in[18], *pb = (const float*)d_in[19];
  const float *sw = (const float*)d_in[20], *sb = (const float*)d_in[21];
  const float* act_emb = (const float*)d_in[22];
  const float *rw = (const float*)d_in[23], *rb = (const float*)d_in[24];
  const float *hw1 = (const float*)d_in[61], *hb1 = (const float*)d_in[62];
  const float *hw2 = (const float*)d_in[63], *hb2 = (const float*)d_in[64];

  char* wp = (char*)d_ws;
  auto carve = [&](size_t bytes) -> void* {
    void* p = (void*)wp;
    wp += (bytes + 255) & ~(size_t)255;
    return p;
  };
  __bf16* root_h = (__bf16*)carve((size_t)ROWS * 128 * 2);
  __bf16* attr_h = (__bf16*)carve((size_t)ROWS * 128 * 2);
  __bf16* state  = (__bf16*)carve((size_t)ROWS * 256 * 2);
  float*  x      = (float*)carve((size_t)SEQS * SLEN * 128 * 4);
  __bf16* qkv    = (__bf16*)carve((size_t)SEQS * SLEN * 384 * 2);
  __bf16* obuf   = (__bf16*)carve((size_t)SEQS * SLEN * 128 * 2);
  __bf16 *w1t[3], *w2t[3];
  for (int L = 0; L < 3; ++L) {
    w1t[L] = (__bf16*)carve((size_t)DFFD * 128 * 2);
    w2t[L] = (__bf16*)carve((size_t)128 * DFFD * 2);
  }

  (void)hipFuncSetAttribute((const void*)k_tree, hipFuncAttributeMaxDynamicSharedMemorySize, 286592);
  (void)hipFuncSetAttribute((const void*)k_attr, hipFuncAttributeMaxDynamicSharedMemorySize, 180224);
  (void)hipFuncSetAttribute((const void*)k_proj, hipFuncAttributeMaxDynamicSharedMemorySize, 131072);
  (void)hipFuncSetAttribute((const void*)k_s, hipFuncAttributeMaxDynamicSharedMemorySize, 65536);
  (void)hipFuncSetAttribute((const void*)k_qkv, hipFuncAttributeMaxDynamicSharedMemorySize, 98304);
  (void)hipFuncSetAttribute((const void*)k_attn, hipFuncAttributeMaxDynamicSharedMemorySize, 233472);
  (void)hipFuncSetAttribute((const void*)k_ff, hipFuncAttributeMaxDynamicSharedMemorySize, 73728);
  (void)hipFuncSetAttribute((const void*)k_head, hipFuncAttributeMaxDynamicSharedMemorySize, 102400);

  for (int L = 0; L < 3; ++L) {
    const float* fw1 = (const float*)d_in[25 + 12 * L + 6];
    const float* fw2 = (const float*)d_in[25 + 12 * L + 8];
    k_prep<<<1024, 256, 0, stream>>>(fw1, w1t[L], 128, DFFD, 128);
    k_prep<<<1024, 256, 0, stream>>>(fw2, w2t[L], DFFD, 128, DFFD);
  }
  k_tree<<<2048, 256, 286592, stream>>>(forest, tw_iou, tb_iou, tu_iou, tw_f, tb_f, tu_f, root_h);
  k_attr<<<128, 256, 180224, stream>>>(aattr, aw1, ab1, aw2, ab2, attr_h);
  k_proj<<<128, 256, 131072, stream>>>(root_h, attr_h, pw, pb, state);
  k_s<<<128, 256, 65536, stream>>>(state, sw, sb, x);
  k_ra<<<8192, 256, 0, stream>>>(rtgs, actions, act_emb, rw, rb, x);
  for (int L = 0; L < 3; ++L) {
    const float* in_w  = (const float*)d_in[25 + 12 * L + 0];
    const float* in_b  = (const float*)d_in[25 + 12 * L + 1];
    const float* out_w = (const float*)d_in[25 + 12 * L + 2];
    const float* out_b = (const float*)d_in[25 + 12 * L + 3];
    const float* ln1g  = (const float*)d_in[25 + 12 * L + 4];
    const float* ln1b  = (const float*)d_in[25 + 12 * L + 5];
    const float* ff_b1 = (const float*)d_in[25 + 12 * L + 7];
    const float* ff_b2 = (const float*)d_in[25 + 12 * L + 9];
    const float* ln2g  = (const float*)d_in[25 + 12 * L + 10];
    const float* ln2b  = (const float*)d_in[25 + 12 * L + 11];
    k_qkv<<<384, 256, 98304, stream>>>(x, in_w, in_b, qkv);
    k_attn<<<1024, 256, 233472, stream>>>(qkv, obuf);
    k_oln<<<384, 256, 32768, stream>>>(obuf, out_w, out_b, ln1g, ln1b, x);
    k_ff<<<3072, 256, 73728, stream>>>(x, w1t[L], ff_b1, w2t[L], ff_b2, ln2g, ln2b);
  }
  k_head<<<1, 256, 102400, stream>>>(x, hw1, hb1, hw2, hb2, (float*)d_out);
}
